// MambaBlock_6107443495369
// MI455X (gfx1250) — matrix operand order repaired and checked by reading, not compiled, not run on hardware
//
#include <hip/hip_runtime.h>

// ---------------------------------------------------------------------------
// Mamba block for MI455X (gfx1250, wave32, WMMA).
// Big GEMMs: bf16 WMMA, register-blocked 2x4 per wave, A-tile staged in LDS
//            via double-buffered GLOBAL_LOAD_ASYNC_TO_LDS (ASYNCcnt).
// Scan: fp32, parallel over (batch, channel, state) with shfl_xor reduction.
// ---------------------------------------------------------------------------

typedef unsigned short u16;
typedef unsigned int   u32;

typedef __bf16 bf16_t;
typedef bf16_t v16bf __attribute__((ext_vector_type(16)));
typedef float  v8f   __attribute__((ext_vector_type(8)));

union ABFrag { uint4 q[2]; v16bf v; };

// problem constants
constexpr int BATCH = 4;
constexpr int SEQ   = 2048;
constexpr int DM    = 1024;
constexpr int DI    = 2048;
constexpr int DTR   = 64;
constexpr int NST   = 16;
constexpr int NXP   = DTR + 2 * NST;     // 96
constexpr int T     = BATCH * SEQ;       // 8192

__device__ __forceinline__ u16 f2bf(float f) {
  u32 u = __float_as_uint(f);
  u32 r = (u + 0x7FFFu + ((u >> 16) & 1u)) >> 16;
  return (u16)r;
}

__device__ __forceinline__ float softplusf(float x) {
  return x > 20.f ? x : log1pf(__expf(x));
}

__device__ __forceinline__ float siluf(float x) {
  return x / (1.f + __expf(-x));
}

// --- CDNA5 async copy: global -> LDS, 8 bytes per lane (ASYNCcnt) ----------
__device__ __forceinline__ void async_copy_b64(u32 lds_off, const void* gaddr) {
  asm volatile("global_load_async_to_lds_b64 %0, %1, off"
               :: "v"(lds_off), "v"(gaddr) : "memory");
}

__device__ __forceinline__ void wait_async0() {
#if __has_builtin(__builtin_amdgcn_s_wait_asynccnt)
  __builtin_amdgcn_s_wait_asynccnt(0);
#else
  asm volatile("s_wait_asynccnt 0x0" ::: "memory");
#endif
}

// ---------------------------------------------------------------------------
// fp32 -> bf16 elementwise (weight conversion)
// ---------------------------------------------------------------------------
__global__ __launch_bounds__(256) void cvt_bf16_kernel(const float* __restrict__ src,
                                                       u16* __restrict__ dst, int n) {
  int i = blockIdx.x * 256 + threadIdx.x;
  if (i < n) dst[i] = f2bf(src[i]);
}

// ---------------------------------------------------------------------------
// LayerNorm over D_MODEL per token, bf16 output. One block (256 thr) per token.
// ---------------------------------------------------------------------------
__global__ __launch_bounds__(256) void ln_kernel(const float* __restrict__ x,
                                                 const float* __restrict__ g,
                                                 const float* __restrict__ b,
                                                 u16* __restrict__ xn) {
  __shared__ float red[256];
  int t = blockIdx.x;
  const float* row = x + (size_t)t * DM;

  float s = 0.f;
  for (int i = threadIdx.x; i < DM; i += 256) s += row[i];
  red[threadIdx.x] = s;
  __syncthreads();
  for (int o = 128; o > 0; o >>= 1) {
    if (threadIdx.x < o) red[threadIdx.x] += red[threadIdx.x + o];
    __syncthreads();
  }
  float mean = red[0] * (1.f / DM);
  __syncthreads();

  float vs = 0.f;
  for (int i = threadIdx.x; i < DM; i += 256) {
    float d = row[i] - mean;
    vs += d * d;
  }
  red[threadIdx.x] = vs;
  __syncthreads();
  for (int o = 128; o > 0; o >>= 1) {
    if (threadIdx.x < o) red[threadIdx.x] += red[threadIdx.x + o];
    __syncthreads();
  }
  float rstd = rsqrtf(red[0] * (1.f / DM) + 1e-5f);

  for (int i = threadIdx.x; i < DM; i += 256)
    xn[(size_t)t * DM + i] = f2bf((row[i] - mean) * rstd * g[i] + b[i]);
}

// ---------------------------------------------------------------------------
// Shared epilogue for the WMMA GEMMs.
//   MODE 0: in_proj  -> split into x_in / z        (f32)
//   MODE 1: x_proj   -> xdbl f32 ; dt slice -> bf16 copy
//   MODE 2: dt_proj  -> softplus(acc + bias) -> delta f32
//   MODE 3: out_proj -> out = x_res + acc * exp(-softplus(log_decay)*tau_h)
// ---------------------------------------------------------------------------
template <int MODE>
__device__ __forceinline__ void gemm_epilogue(
    float v, int gm, int gn,
    float* __restrict__ out0, float* __restrict__ out1, u16* __restrict__ outb,
    const float* __restrict__ aux0, const float* __restrict__ aux1,
    const float* __restrict__ aux2) {
  if (MODE == 0) {
    if (gn < DI) out0[(size_t)gm * DI + gn] = v;
    else         out1[(size_t)gm * DI + (gn - DI)] = v;
  } else if (MODE == 1) {
    out0[(size_t)gm * NXP + gn] = v;
    if (gn < DTR) outb[(size_t)gm * DTR + gn] = f2bf(v);
  } else if (MODE == 2) {
    out0[(size_t)gm * DI + gn] = softplusf(v + aux0[gn]);
  } else { // MODE 3
    float tau_h = fmaxf(aux1[gm] * (1.f / 3600.f), 1e-4f);
    float gate  = __expf(-softplusf(aux0[gn]) * tau_h);
    out0[(size_t)gm * DM + gn] = aux2[(size_t)gm * DM + gn] + v * gate;
  }
}

// ---------------------------------------------------------------------------
// Big-GEMM kernel: C[M,N] = A[M,K](bf16,row) * W[N,K](bf16,row)^T, fp32 accum.
// Block = 8 waves. Block tile: 32 rows x 512 cols. Each wave: 2x4 16x16 tiles
// (8 accumulators). The 32x32 bf16 A-tile per k-step is staged in LDS with
// double-buffered global_load_async_to_lds_b64 and shared by all 8 waves;
// fragments are read back with ds_load_b128 (80B row pitch, conflict-spread).
// Requires N % 512 == 0, M % 32 == 0, K % 32 == 0. Grid exact, EXEC all-ones.
// ---------------------------------------------------------------------------
template <int MODE>
__global__ __launch_bounds__(256) void gemm_wmma_lds(
    const u16* __restrict__ A, const u16* __restrict__ W,
    int M, int N, int K,
    float* __restrict__ out0, float* __restrict__ out1, u16* __restrict__ outb,
    const float* __restrict__ aux0, const float* __restrict__ aux1,
    const float* __restrict__ aux2) {
  constexpr int MB = 2, NB = 4;
  constexpr int ROWP = 80;                 // LDS row pitch (bytes) for 64B rows
  constexpr int BUFB = 32 * ROWP;          // 2560B per buffer
  __shared__ __align__(16) char lds[2 * BUFB];

  int tid  = threadIdx.x;
  int w    = tid >> 5;
  int lane = tid & 31;
  int mn    = lane & 15;
  int khalf = lane >> 4;

  int nBlocksN = N >> 9;                   // N / 512
  int bm = blockIdx.x / nBlocksN;
  int bn = blockIdx.x - bm * nBlocksN;

  // async-copy assignment: 256 threads x 8B = 2KB = 32 rows x 64 bytes
  int crow = tid >> 3;                     // 0..31
  int cchk = tid & 7;                      // 0..7
  const char* gsrc = (const char*)(A + (size_t)(bm * 32 + crow) * K) + cchk * 8;
  u32 ldsbase = (u32)(uintptr_t)&lds[0];
  u32 ldst    = ldsbase + crow * ROWP + cchk * 8;

  // B columns: weight rows, contiguous in K
  int ntbase = (bn * 8 + w) * NB;
  const u16* bcol[NB];
#pragma unroll
  for (int nj = 0; nj < NB; ++nj)
    bcol[nj] = W + (size_t)((ntbase + nj) * 16 + mn) * K;

  v8f acc[MB][NB];
#pragma unroll
  for (int mi = 0; mi < MB; ++mi)
#pragma unroll
    for (int nj = 0; nj < NB; ++nj)
#pragma unroll
      for (int r = 0; r < 8; ++r) acc[mi][nj][r] = 0.f;

  int KS = K >> 5;
  async_copy_b64(ldst, gsrc);              // prologue: k-step 0 -> buffer 0
  for (int ks = 0; ks < KS; ++ks) {
    wait_async0();                         // copy into buf[ks&1] complete
    __syncthreads();                       // visible to all waves; prev buf free
    if (ks + 1 < KS)
      async_copy_b64(ldst + ((ks + 1) & 1) * BUFB, gsrc + (size_t)(ks + 1) * 64);

    const char* lb = lds + (ks & 1) * BUFB;
    ABFrag af[MB], bf[NB];
#pragma unroll
    for (int mi = 0; mi < MB; ++mi) {
      const char* lr = lb + (mi * 16 + mn) * ROWP;
      af[mi].q[0] = *(const uint4*)(lr + khalf * 16);
      af[mi].q[1] = *(const uint4*)(lr + 32 + khalf * 16);
    }
#pragma unroll
    for (int nj = 0; nj < NB; ++nj) {
      const u16* bp = bcol[nj] + ks * 32 + khalf * 8;
      bf[nj].q[0] = *(const uint4*)(bp);
      bf[nj].q[1] = *(const uint4*)(bp + 16);
    }
#pragma unroll
    for (int mi = 0; mi < MB; ++mi)
#pragma unroll
      for (int nj = 0; nj < NB; ++nj)
        acc[mi][nj] = __builtin_amdgcn_wmma_f32_16x16x32_bf16(
            false, af[mi].v, false, bf[nj].v, (short)0, acc[mi][nj],
            false, false);
  }

#pragma unroll
  for (int mi = 0; mi < MB; ++mi)
#pragma unroll
    for (int nj = 0; nj < NB; ++nj) {
      int gn = (ntbase + nj) * 16 + mn;
#pragma unroll
      for (int r = 0; r < 8; ++r) {
        int gm = bm * 32 + mi * 16 + r + 8 * khalf;
        gemm_epilogue<MODE>(acc[mi][nj][r], gm, gn, out0, out1, outb,
                            aux0, aux1, aux2);
      }
    }
}

// ---------------------------------------------------------------------------
// Flat WMMA GEMM (one 16x16 tile per wave) for the narrow x_proj (N=96).
// ---------------------------------------------------------------------------
template <int MODE>
__global__ __launch_bounds__(256) void gemm_wmma_flat(
    const u16* __restrict__ A, const u16* __restrict__ W,
    int M, int N, int K,
    float* __restrict__ out0, float* __restrict__ out1, u16* __restrict__ outb,
    const float* __restrict__ aux0, const float* __restrict__ aux1,
    const float* __restrict__ aux2) {
  int wave   = (blockIdx.x * 256 + threadIdx.x) >> 5;
  int lane   = threadIdx.x & 31;
  int tilesN = N >> 4;
  int total  = (M >> 4) * tilesN;
  if (wave >= total) return;               // uniform per wave

  int tm = wave / tilesN;
  int tn = wave - tm * tilesN;
  int mn    = lane & 15;
  int khalf = lane >> 4;

  const u16* arow = A + (size_t)(tm * 16 + mn) * K;
  const u16* bcol = W + (size_t)(tn * 16 + mn) * K;

  v8f acc = {0.f, 0.f, 0.f, 0.f, 0.f, 0.f, 0.f, 0.f};
  for (int k0 = 0; k0 < K; k0 += 32) {
    ABFrag af, bf;
    af.q[0] = *(const uint4*)(arow + k0 + khalf * 8);
    af.q[1] = *(const uint4*)(arow + k0 + 16 + khalf * 8);
    bf.q[0] = *(const uint4*)(bcol + k0 + khalf * 8);
    bf.q[1] = *(const uint4*)(bcol + k0 + 16 + khalf * 8);
    acc = __builtin_amdgcn_wmma_f32_16x16x32_bf16(
        false, af.v, false, bf.v, (short)0, acc, false, false);
  }

  int gn = tn * 16 + mn;
#pragma unroll
  for (int r = 0; r < 8; ++r) {
    int gm = tm * 16 + r + 8 * khalf;
    gemm_epilogue<MODE>(acc[r], gm, gn, out0, out1, outb, aux0, aux1, aux2);
  }
}

// ---------------------------------------------------------------------------
// Causal depthwise conv (K=4) + bias + SiLU. One thread per (t, c).
// ---------------------------------------------------------------------------
__global__ __launch_bounds__(256) void conv_silu_kernel(
    const float* __restrict__ xin, const float* __restrict__ w,
    const float* __restrict__ bias, float* __restrict__ u,
    u16* __restrict__ ub) {
  int idx = blockIdx.x * 256 + threadIdx.x;
  if (idx >= T * DI) return;
  int c = idx & (DI - 1);
  int t = idx / DI;
  int l = t & (SEQ - 1);

  float acc = bias[c];
  const float* wc = w + c * 4;
#pragma unroll
  for (int j = 0; j < 4; ++j) {
    int lt = l - 3 + j;
    if (lt >= 0) acc += wc[j] * xin[(size_t)(t - 3 + j) * DI + c];
  }
  float s = siluf(acc);
  u[idx]  = s;
  ub[idx] = f2bf(s);
}

// ---------------------------------------------------------------------------
// Selective scan: one thread per (batch, channel, state); shfl_xor butterfly
// reduces the 16 per-state contributions each timestep (wave32 lane perms).
// ---------------------------------------------------------------------------
__global__ __launch_bounds__(256) void scan_kernel(
    const float* __restrict__ delta, const float* __restrict__ u,
    const float* __restrict__ xdbl,  const float* __restrict__ z,
    const float* __restrict__ A_log, const float* __restrict__ Dp,
    u16* __restrict__ ybf) {
  int g     = blockIdx.x * 256 + threadIdx.x;   // [0, BATCH*DI*NST)
  int batch = g / (DI * NST);
  int rem   = g - batch * (DI * NST);
  int d     = rem >> 4;
  int n     = rem & 15;

  float a  = -__expf(A_log[d * NST + n]);
  float Dv = Dp[d];
  float h  = 0.f;
  size_t tb = (size_t)batch * SEQ;

  for (int t = 0; t < SEQ; ++t) {
    size_t row = tb + t;
    float dt = delta[row * DI + d];
    float ut = u[row * DI + d];
    float Bn = xdbl[row * NXP + DTR + n];
    float Cn = xdbl[row * NXP + DTR + NST + n];

    h = __expf(dt * a) * h + (dt * ut) * Bn;
    float p = h * Cn;
    p += __shfl_xor(p, 1, 16);
    p += __shfl_xor(p, 2, 16);
    p += __shfl_xor(p, 4, 16);
    p += __shfl_xor(p, 8, 16);

    if (n == 0) {
      float y  = p + ut * Dv;
      float zt = z[row * DI + d];
      ybf[row * DI + d] = f2bf(y * siluf(zt));
    }
  }
}

// ---------------------------------------------------------------------------
// Launch
// ---------------------------------------------------------------------------
extern "C" void kernel_launch(void* const* d_in, const int* in_sizes, int n_in,
                              void* d_out, int out_size, void* d_ws, size_t ws_size,
                              hipStream_t stream) {
  const float* x          = (const float*)d_in[0];
  const float* tau        = (const float*)d_in[1];
  const float* ln_g       = (const float*)d_in[2];
  const float* ln_b       = (const float*)d_in[3];
  const float* in_proj_w  = (const float*)d_in[4];
  const float* conv_w     = (const float*)d_in[5];
  const float* conv_b     = (const float*)d_in[6];
  const float* x_proj_w   = (const float*)d_in[7];
  const float* dt_proj_w  = (const float*)d_in[8];
  const float* dt_proj_b  = (const float*)d_in[9];
  const float* A_log      = (const float*)d_in[10];
  const float* Dp         = (const float*)d_in[11];
  const float* out_proj_w = (const float*)d_in[12];
  const float* log_decay  = (const float*)d_in[13];
  float* out = (float*)d_out;

  char* ws = (char*)d_ws;
  size_t off = 0;
  auto alloc = [&](size_t bytes) -> char* {
    char* p = ws + off;
    off += (bytes + 255) & ~(size_t)255;
    return p;
  };

  u16*   xn    = (u16*)alloc((size_t)T * DM * 2);        // LN output bf16
  u16*   Win   = (u16*)alloc((size_t)2 * DI * DM * 2);   // in_proj bf16
  u16*   Wx    = (u16*)alloc((size_t)NXP * DI * 2);      // x_proj bf16
  u16*   Wdt   = (u16*)alloc((size_t)DI * DTR * 2);      // dt_proj bf16
  u16*   Wout  = (u16*)alloc((size_t)DM * DI * 2);       // out_proj bf16
  float* xin   = (float*)alloc((size_t)T * DI * 4);      // conv branch (f32)
  float* zbuf  = (float*)alloc((size_t)T * DI * 4);      // gate branch (f32)
  float* ubuf  = (float*)alloc((size_t)T * DI * 4);      // conv+silu out f32
  u16*   ubf   = (u16*)alloc((size_t)T * DI * 2);        // conv+silu out bf16
  float* xdbl  = (float*)alloc((size_t)T * NXP * 4);     // dt|B|C f32
  u16*   dtbf  = (u16*)alloc((size_t)T * DTR * 2);       // dt slice bf16
  u16*   ybf   = (u16*)alloc((size_t)T * DI * 2);        // gated scan out bf16
  float* deltb = xin;                                    // reuse: xin dead after conv

  // weight conversions
  cvt_bf16_kernel<<<(2 * DI * DM + 255) / 256, 256, 0, stream>>>(in_proj_w, Win, 2 * DI * DM);
  cvt_bf16_kernel<<<(NXP * DI + 255) / 256, 256, 0, stream>>>(x_proj_w, Wx, NXP * DI);
  cvt_bf16_kernel<<<(DI * DTR + 255) / 256, 256, 0, stream>>>(dt_proj_w, Wdt, DI * DTR);
  cvt_bf16_kernel<<<(DM * DI + 255) / 256, 256, 0, stream>>>(out_proj_w, Wout, DM * DI);

  // LayerNorm
  ln_kernel<<<T, 256, 0, stream>>>(x, ln_g, ln_b, xn);

  // in_proj: [T,1024] x [4096,1024]^T -> x_in | z   (LDS-staged WMMA)
  gemm_wmma_lds<0><<<(T / 32) * (2 * DI / 512), 256, 0, stream>>>(
      xn, Win, T, 2 * DI, DM, xin, zbuf, nullptr, nullptr, nullptr, nullptr);

  // depthwise causal conv + SiLU
  conv_silu_kernel<<<(T * DI + 255) / 256, 256, 0, stream>>>(xin, conv_w, conv_b, ubuf, ubf);

  // x_proj: [T,2048] x [96,2048]^T -> xdbl (+ bf16 dt slice)   (flat WMMA)
  {
    int tiles = (T / 16) * (NXP / 16);
    gemm_wmma_flat<1><<<(tiles + 7) / 8, 256, 0, stream>>>(
        ubf, Wx, T, NXP, DI, xdbl, nullptr, dtbf, nullptr, nullptr, nullptr);
  }

  // dt_proj + softplus -> delta (reuses xin buffer)   (LDS-staged WMMA)
  gemm_wmma_lds<2><<<(T / 32) * (DI / 512), 256, 0, stream>>>(
      dtbf, Wdt, T, DI, DTR, deltb, nullptr, nullptr, dt_proj_b, nullptr, nullptr);

  // selective scan + D skip + SiLU(z) gate -> ybf
  scan_kernel<<<(BATCH * DI * NST) / 256, 256, 0, stream>>>(
      deltb, ubuf, xdbl, zbuf, A_log, Dp, ybf);

  // out_proj + ZOH decay gate + residual -> d_out   (LDS-staged WMMA)
  gemm_wmma_lds<3><<<(T / 32) * (DM / 512), 256, 0, stream>>>(
      ybf, Wout, T, DM, DI, out, nullptr, nullptr, log_decay, tau, x);
}
